// TestModel_6_4355096838967
// MI455X (gfx1250) — matrix operand order repaired and checked by reading, not compiled, not run on hardware
//
#include <hip/hip_runtime.h>
#include <hip/hip_bf16.h>

// ---------------------------------------------------------------------------
// UniMol encoder for MI455X (gfx1250, wave32, WMMA).
//  - all dense GEMMs: v_wmma_f32_16x16x32_f16, 64x32 tile per wave (8 WMMA /
//    6 fragment loads), pointer-walk addressing (contiguous packed k-tiles)
//  - weights pre-packed to WMMA B-fragment layout (f16, b128 loads)
//  - f16 activations stored directly in WMMA A-fragment layout by producers
//  - fused scores+softmax (one pass over the [B,H,N,N] pair tensor)
//  - K/V transposed to [B,H,DH,N]; probs kept f16
// ---------------------------------------------------------------------------

constexpr int L_ = 15, D_ = 512, F_ = 2048, H_ = 64, K_ = 128, DH_ = 8;
constexpr int B_ = 8, N_ = 128;

typedef __attribute__((ext_vector_type(16))) _Float16 v16h;
typedef __attribute__((ext_vector_type(8)))  float    v8f;

// ---- packed-fragment address maps (CDNA5 ISA 7.12.2, wave32) ---------------

// A-matrix 16x32 f16 tile: lane = (m&15) + 16*hi ; element e <-> K per table.
__device__ __forceinline__ size_t pa_off(int m, int k, int Kt) {
    const int kl = k & 31;
    const int hi = (kl >> 3) & 1;            // {8..15,24..31} live in lanes 16-31
    const int km = kl - 8 * hi;              // in {0..7} u {16..23}
    const int r  = (km < 8) ? (km >> 1) : (4 + ((km - 16) >> 1));
    const int e  = 2 * r + (km & 1);
    const int lane = (m & 15) + 16 * hi;
    return ((((size_t)(m >> 4) * (Kt >> 5) + (k >> 5)) * 32 + lane) << 4) + e;
}

// fp32 row-major [Kt,Nn] weight -> packed f16 B-fragment tiles
// B tile: lane = (n&15) + 16*hi ; element e holds K = 16*(e>>3) + 8*hi + (e&7).
__global__ void pack_b_kernel(const float* W, _Float16* P, int Kt, int Nn) {
    const int idx  = blockIdx.x * blockDim.x + threadIdx.x;   // Kt*Nn halves
    const int e    = idx & 15;
    const int lane = (idx >> 4) & 31;
    const int t    = idx >> 9;
    const int tn_c = Nn >> 4;
    const int tk = t / tn_c, tn = t % tn_c;
    const int n = (tn << 4) + (lane & 15);
    const int k = (tk << 5) + ((e >> 3) << 4) + ((lane >> 4) << 3) + (e & 7);
    P[idx] = (_Float16)W[(size_t)k * Nn + n];
}

// ---- WMMA inner loop: 64x32 wave tile, pointer-walk over contiguous k-tiles

__device__ __forceinline__ void wmma_accum4x2(const _Float16* A, const _Float16* Bp,
                                              int Kt, int Nn, int m0, int n0,
                                              v8f acc[4][2]) {
    const int lane   = threadIdx.x & 31;
    const int ktiles = Kt >> 5;
    const _Float16* ap = A + (((size_t)(m0 >> 4) * ktiles) * 32 + lane) * 16;
    const _Float16* bp = Bp + (((size_t)(n0 >> 4)) * 32 + lane) * 16;
    const size_t amstep = (size_t)ktiles * 512;     // halves between m-tiles
    const size_t bkstep = (size_t)(Nn >> 4) * 512;  // halves between k-tiles (B)
    for (int kt = 0; kt < ktiles; ++kt) {
        __builtin_prefetch(bp + bkstep, 0, 1);      // global_prefetch_b8
        v16h b0 = *(const v16h*)bp;
        v16h b1 = *(const v16h*)(bp + 512);
#pragma unroll
        for (int t = 0; t < 4; ++t) {
            v16h a = *(const v16h*)(ap + (size_t)t * amstep);
            acc[t][0] = __builtin_amdgcn_wmma_f32_16x16x32_f16(
                false, a, false, b0, (short)0, acc[t][0], false, false);
            acc[t][1] = __builtin_amdgcn_wmma_f32_16x16x32_f16(
                false, a, false, b1, (short)0, acc[t][1], false, false);
        }
        ap += 512;                                  // next k-tile of A (contig)
        bp += bkstep;
    }
}

__device__ __forceinline__ float gelu_exact(float x) {
    return 0.5f * x * (1.0f + erff(x * 0.70710678118654752f));
}

// ---- GEMM kernels: wave = 64-row group x 32-col group; exact grids ---------

__global__ void gemm_f32_kernel(const _Float16* A, const _Float16* Bp, const float* bias,
                                float* C, int Nn, int Kt, float scale) {
    const int wave = blockIdx.x * (blockDim.x >> 5) + (threadIdx.x >> 5);
    const int tn2  = Nn >> 5;
    const int m0   = (wave / tn2) << 6;
    const int n0   = (wave % tn2) << 5;
    v8f acc[4][2] = {};
    wmma_accum4x2(A, Bp, Kt, Nn, m0, n0, acc);
    const int lane = threadIdx.x & 31;
    const int hi   = lane >> 4;
#pragma unroll
    for (int c = 0; c < 2; ++c) {
        const int n    = n0 + (c << 4) + (lane & 15);
        const float bn = bias[n];
#pragma unroll
        for (int t = 0; t < 4; ++t)
#pragma unroll
            for (int r = 0; r < 8; ++r) {
                const int m = m0 + 16 * t + r + 8 * hi;
                C[(size_t)m * Nn + n] = (acc[t][c][r] + bn) * scale;
            }
    }
}

// K/V projection with transposed store -> out[B,H,DH,N]  (Nn==D_, rows = B*N)
__global__ void gemm_kv_perm_kernel(const _Float16* A, const _Float16* Bp,
                                    const float* bias, float* out) {
    const int wave = blockIdx.x * (blockDim.x >> 5) + (threadIdx.x >> 5);
    const int tn2  = D_ >> 5;
    const int m0   = (wave / tn2) << 6;
    const int n0   = (wave % tn2) << 5;
    v8f acc[4][2] = {};
    wmma_accum4x2(A, Bp, D_, D_, m0, n0, acc);
    const int lane = threadIdx.x & 31;
    const int hi   = lane >> 4;
#pragma unroll
    for (int c = 0; c < 2; ++c) {
        const int n = n0 + (c << 4) + (lane & 15);     // n = h*8 + d
        const int h = n >> 3, d = n & 7;
        const float bn = bias[n];
#pragma unroll
        for (int t = 0; t < 4; ++t)
#pragma unroll
            for (int r = 0; r < 8; ++r) {
                const int m = m0 + 16 * t + r + 8 * hi;   // m = b*N + j
                const int b = m >> 7, j = m & 127;
                out[(((size_t)(b * H_ + h)) * DH_ + d) * N_ + j] = acc[t][c][r] + bn;
            }
    }
}

// gelu epilogue, f16 output written in packed-A layout (consumer Kt == Nn)
__global__ void gemm_gelu_pa_kernel(const _Float16* A, const _Float16* Bp,
                                    const float* bias, _Float16* C, int Nn, int Kt) {
    const int wave = blockIdx.x * (blockDim.x >> 5) + (threadIdx.x >> 5);
    const int tn2  = Nn >> 5;
    const int m0   = (wave / tn2) << 6;
    const int n0   = (wave % tn2) << 5;
    v8f acc[4][2] = {};
    wmma_accum4x2(A, Bp, Kt, Nn, m0, n0, acc);
    const int lane = threadIdx.x & 31;
    const int hi   = lane >> 4;
#pragma unroll
    for (int c = 0; c < 2; ++c) {
        const int n    = n0 + (c << 4) + (lane & 15);
        const float bn = bias[n];
#pragma unroll
        for (int t = 0; t < 4; ++t)
#pragma unroll
            for (int r = 0; r < 8; ++r) {
                const int m = m0 + 16 * t + r + 8 * hi;
                C[pa_off(m, n, Nn)] = (_Float16)gelu_exact(acc[t][c][r] + bn);
            }
    }
}

// fp32 in-place residual epilogue:  C += A@W + bias
__global__ void gemm_res_f32_kernel(const _Float16* A, const _Float16* Bp,
                                    const float* bias, float* C, int Nn, int Kt) {
    const int wave = blockIdx.x * (blockDim.x >> 5) + (threadIdx.x >> 5);
    const int tn2  = Nn >> 5;
    const int m0   = (wave / tn2) << 6;
    const int n0   = (wave % tn2) << 5;
    v8f acc[4][2] = {};
    wmma_accum4x2(A, Bp, Kt, Nn, m0, n0, acc);
    const int lane = threadIdx.x & 31;
    const int hi   = lane >> 4;
#pragma unroll
    for (int c = 0; c < 2; ++c) {
        const int n    = n0 + (c << 4) + (lane & 15);
        const float bn = bias[n];
#pragma unroll
        for (int t = 0; t < 4; ++t)
#pragma unroll
            for (int r = 0; r < 8; ++r) {
                const int m = m0 + 16 * t + r + 8 * hi;
                const size_t idx = (size_t)m * Nn + n;
                C[idx] = C[idx] + acc[t][c][r] + bn;
            }
    }
}

// hproj @ proj_w2 with permuted store -> bias0[b,h,i,j]   (Nn==H_, Kt==K_)
__global__ void gemm_biasperm_kernel(const _Float16* A, const _Float16* Bp,
                                     const float* bias, float* out) {
    const int wave = blockIdx.x * (blockDim.x >> 5) + (threadIdx.x >> 5);
    const int tn2  = H_ >> 5;
    const int m0   = (wave / tn2) << 6;
    const int n0   = (wave % tn2) << 5;
    v8f acc[4][2] = {};
    wmma_accum4x2(A, Bp, K_, H_, m0, n0, acc);
    const int lane = threadIdx.x & 31;
    const int hi   = lane >> 4;
#pragma unroll
    for (int c = 0; c < 2; ++c) {
        const int h    = n0 + (c << 4) + (lane & 15);
        const float bn = bias[h];
#pragma unroll
        for (int t = 0; t < 4; ++t)
#pragma unroll
            for (int r = 0; r < 8; ++r) {
                const int m  = m0 + 16 * t + r + 8 * hi;   // (b*N + i)*N + j
                const int b  = m >> 14;
                const int ij = m & 16383;
                const int i  = ij >> 7, j = ij & 127;
                out[(((size_t)b * H_ + h) * N_ + i) * N_ + j] = acc[t][c][r] + bn;
            }
    }
}

// ---- Gaussian pair features -> g (f16, packed-A layout, Kt=K_) -------------

__global__ void gauss_kernel(const float* dist, const int* etype, const float* gmul,
                             const float* gbias, const float* means, const float* stds,
                             _Float16* g) {
    const int idx  = blockIdx.x * blockDim.x + threadIdx.x;   // B*N*N*K
    const int kk   = idx & (K_ - 1);
    const int pair = idx >> 7;
    const int et   = etype[pair];
    const float x  = gmul[et] * dist[pair] + gbias[et];
    const float sd = fabsf(stds[kk]) + 1e-5f;
    const float z  = (x - means[kk]) / sd;
    g[pa_off(pair, kk, K_)] = (_Float16)(expf(-0.5f * z * z) / (2.50662607f * sd));
}

// ---- LayerNorm (wave32 shuffle; one wave per D=512 row) --------------------

__global__ void ln_pa_kernel(const float* x, const float* s, const float* b,
                             _Float16* y) {   // f16 out in packed-A layout (Kt=D_)
    const int lane = threadIdx.x & 31;
    const int row  = blockIdx.x * (blockDim.x >> 5) + (threadIdx.x >> 5);
    const float* src = x + (size_t)row * D_;
    float v[16], sum = 0.f, sq = 0.f;
#pragma unroll
    for (int t = 0; t < 16; ++t) {
        v[t] = src[t * 32 + lane];
        sum += v[t]; sq += v[t] * v[t];
    }
#pragma unroll
    for (int m = 16; m >= 1; m >>= 1) {
        sum += __shfl_xor(sum, m, 32);
        sq  += __shfl_xor(sq,  m, 32);
    }
    const float mean = sum * (1.0f / D_);
    const float inv  = rsqrtf(sq * (1.0f / D_) - mean * mean + 1e-5f);
#pragma unroll
    for (int t = 0; t < 16; ++t) {
        const int c = t * 32 + lane;
        y[pa_off(row, c, D_)] = (_Float16)((v[t] - mean) * inv * s[c] + b[c]);
    }
}

__global__ void ln_f32_kernel(const float* x, const float* s, const float* b, float* y) {
    const int lane = threadIdx.x & 31;
    const int row  = blockIdx.x * (blockDim.x >> 5) + (threadIdx.x >> 5);
    const float* src = x + (size_t)row * D_;
    float v[16], sum = 0.f, sq = 0.f;
#pragma unroll
    for (int t = 0; t < 16; ++t) {
        v[t] = src[t * 32 + lane];
        sum += v[t]; sq += v[t] * v[t];
    }
#pragma unroll
    for (int m = 16; m >= 1; m >>= 1) {
        sum += __shfl_xor(sum, m, 32);
        sq  += __shfl_xor(sq,  m, 32);
    }
    const float mean = sum * (1.0f / D_);
    const float inv  = rsqrtf(sq * (1.0f / D_) - mean * mean + 1e-5f);
    float* dst = y + (size_t)row * D_;
#pragma unroll
    for (int t = 0; t < 16; ++t) {
        const int c = t * 32 + lane;
        dst[c] = (v[t] - mean) * inv * s[c] + b[c];
    }
}

__global__ void embed_ln_kernel(const int* toks, const float* emb, const float* s,
                                const float* b, float* y) {
    const int lane = threadIdx.x & 31;
    const int row  = blockIdx.x * (blockDim.x >> 5) + (threadIdx.x >> 5);
    const float* src = emb + (size_t)toks[row] * D_;
    float v[16], sum = 0.f, sq = 0.f;
#pragma unroll
    for (int t = 0; t < 16; ++t) {
        v[t] = src[t * 32 + lane];
        sum += v[t]; sq += v[t] * v[t];
    }
#pragma unroll
    for (int m = 16; m >= 1; m >>= 1) {
        sum += __shfl_xor(sum, m, 32);
        sq  += __shfl_xor(sq,  m, 32);
    }
    const float mean = sum * (1.0f / D_);
    const float inv  = rsqrtf(sq * (1.0f / D_) - mean * mean + 1e-5f);
    float* dst = y + (size_t)row * D_;
#pragma unroll
    for (int t = 0; t < 16; ++t) {
        const int c = t * 32 + lane;
        dst[c] = (v[t] - mean) * inv * s[c] + b[c];
    }
}

// ---- Fused attention scores + softmax (one wave per (b,h,i) row) -----------
// scores = q.kT + bias  (kept fp32 for the next layer's pair bias)
// probs  = softmax(scores) (f16)

__global__ void scores_softmax_kernel(const float* q, const float* kT, const float* bias,
                                      float* scores, _Float16* probs) {
    const int lane = threadIdx.x & 31;
    const int row  = blockIdx.x * (blockDim.x >> 5) + (threadIdx.x >> 5); // (b*H+h)*N+i
    const int i  = row & 127;
    const int bh = row >> 7;              // b*H + h
    const int b  = bh >> 6;
    const int h  = bh & 63;
    const float* qp = q + (size_t)(b * N_ + i) * D_ + h * DH_;
    float qr[DH_];
#pragma unroll
    for (int d = 0; d < DH_; ++d) qr[d] = qp[d];
    const float* kb   = kT + ((size_t)bh * DH_) * N_;
    const float* brow = bias + (size_t)row * N_;
    float v[4], mx = -3.0e38f;
#pragma unroll
    for (int t = 0; t < 4; ++t) {
        const int j = t * 32 + lane;
        float acc = brow[j];
#pragma unroll
        for (int d = 0; d < DH_; ++d) acc += qr[d] * kb[(size_t)d * N_ + j];
        v[t] = acc;
        mx = fmaxf(mx, acc);
    }
    float* srow = scores + (size_t)row * N_;
#pragma unroll
    for (int t = 0; t < 4; ++t) srow[t * 32 + lane] = v[t];
#pragma unroll
    for (int m = 16; m >= 1; m >>= 1) mx = fmaxf(mx, __shfl_xor(mx, m, 32));
    float sum = 0.f;
#pragma unroll
    for (int t = 0; t < 4; ++t) { v[t] = expf(v[t] - mx); sum += v[t]; }
#pragma unroll
    for (int m = 16; m >= 1; m >>= 1) sum += __shfl_xor(sum, m, 32);
    const float inv = 1.0f / sum;
    _Float16* p = probs + (size_t)row * N_;
#pragma unroll
    for (int t = 0; t < 4; ++t) p[t * 32 + lane] = (_Float16)(v[t] * inv);
}

__global__ void attn_out_kernel(const _Float16* probs, const float* vT, _Float16* o) {
    const int idx = blockIdx.x * blockDim.x + threadIdx.x;  // B*N*D
    const int hd  = idx & (D_ - 1);
    const int row = idx >> 9;          // b*N + i
    const int b   = row >> 7;
    const int i   = row & 127;
    const int h   = hd >> 3, d = hd & 7;
    const _Float16* p = probs + (((size_t)b * H_ + h) * N_ + i) * N_;
    const float*   vp = vT + ((size_t)(b * H_ + h) * DH_ + d) * N_;
    float acc = 0.f;
#pragma unroll 4
    for (int j = 0; j < N_; ++j) acc += (float)p[j] * vp[j];
    o[pa_off(row, hd, D_)] = (_Float16)acc;   // packed-A for the Wo GEMM
}

// ---------------------------------------------------------------------------

extern "C" void kernel_launch(void* const* d_in, const int* in_sizes, int n_in,
                              void* d_out, int out_size, void* d_ws, size_t ws_size,
                              hipStream_t stream) {
    const int*   src_tokens    = (const int*)d_in[0];
    const float* src_distance  = (const float*)d_in[1];
    const int*   src_edge_type = (const int*)d_in[2];
    const float* token_embed   = (const float*)d_in[3];
    const float* gbf_means     = (const float*)d_in[4];
    const float* gbf_stds      = (const float*)d_in[5];
    const float* gbf_mul       = (const float*)d_in[6];
    const float* gbf_bias      = (const float*)d_in[7];
    const float* proj_w1       = (const float*)d_in[8];
    const float* proj_b1       = (const float*)d_in[9];
    const float* proj_w2       = (const float*)d_in[10];
    const float* proj_b2       = (const float*)d_in[11];
    const float* emb_ln_s      = (const float*)d_in[12];
    const float* emb_ln_b      = (const float*)d_in[13];
    const float* Wq = (const float*)d_in[14];  const float* bq = (const float*)d_in[15];
    const float* Wk = (const float*)d_in[16];  const float* bk = (const float*)d_in[17];
    const float* Wv = (const float*)d_in[18];  const float* bv = (const float*)d_in[19];
    const float* Wo = (const float*)d_in[20];  const float* bo = (const float*)d_in[21];
    const float* ln1_s = (const float*)d_in[22]; const float* ln1_b = (const float*)d_in[23];
    const float* ln2_s = (const float*)d_in[24]; const float* ln2_b = (const float*)d_in[25];
    const float* Wf1 = (const float*)d_in[26]; const float* bf1 = (const float*)d_in[27];
    const float* Wf2 = (const float*)d_in[28]; const float* bf2 = (const float*)d_in[29];
    const float* fin_ln_s = (const float*)d_in[30];
    const float* fin_ln_b = (const float*)d_in[31];

    // ---- workspace carve-up (~160 MB) ----
    char* wsb = (char*)d_ws;
    size_t off = 0;
    auto carve = [&](size_t bytes) -> void* {
        void* p = wsb + off;
        off = (off + bytes + 255) & ~(size_t)255;
        return p;
    };
    const size_t PAIRS = (size_t)B_ * N_ * N_;                 // 131072
    _Float16* g     = (_Float16*)carve(PAIRS * K_ * 2);        // packed-A
    _Float16* hproj = (_Float16*)carve(PAIRS * K_ * 2);        // packed-A
    float* biasA    = (float*)carve((size_t)B_ * H_ * N_ * N_ * 4);
    float* biasB    = (float*)carve((size_t)B_ * H_ * N_ * N_ * 4);
    _Float16* probs = (_Float16*)carve((size_t)B_ * H_ * N_ * N_ * 2);
    float* x        = (float*)carve((size_t)B_ * N_ * D_ * 4);
    _Float16* y     = (_Float16*)carve((size_t)B_ * N_ * D_ * 2);  // packed-A
    float* qb       = (float*)carve((size_t)B_ * N_ * D_ * 4);
    float* kT       = (float*)carve((size_t)B_ * N_ * D_ * 4);     // [B,H,DH,N]
    float* vT       = (float*)carve((size_t)B_ * N_ * D_ * 4);     // [B,H,DH,N]
    _Float16* ob    = (_Float16*)carve((size_t)B_ * N_ * D_ * 2);  // packed-A
    _Float16* h1    = (_Float16*)carve((size_t)B_ * N_ * F_ * 2);  // packed-A
    // packed weights (per-layer reuse buffers)
    _Float16* pw1 = (_Float16*)carve((size_t)K_ * K_ * 2);
    _Float16* pw2 = (_Float16*)carve((size_t)K_ * H_ * 2);
    _Float16* pq  = (_Float16*)carve((size_t)D_ * D_ * 2);
    _Float16* pk  = (_Float16*)carve((size_t)D_ * D_ * 2);
    _Float16* pv  = (_Float16*)carve((size_t)D_ * D_ * 2);
    _Float16* po  = (_Float16*)carve((size_t)D_ * D_ * 2);
    _Float16* pf1 = (_Float16*)carve((size_t)D_ * F_ * 2);
    _Float16* pf2 = (_Float16*)carve((size_t)F_ * D_ * 2);

    auto gemm_blocks = [](int M, int Nn) { return ((M / 64) * (Nn / 32)) / 4; };
    const int ROWS = B_ * N_;                       // 1024
    const float scale = 0.35355339059327373f;       // DH^-0.5

    // --- Gaussian features + gbf projection -> bias0 (into biasA) ---
    gauss_kernel<<<(int)(PAIRS * K_ / 256), 256, 0, stream>>>(
        src_distance, src_edge_type, gbf_mul, gbf_bias, gbf_means, gbf_stds, g);
    pack_b_kernel<<<(K_ * K_) / 256, 256, 0, stream>>>(proj_w1, pw1, K_, K_);
    pack_b_kernel<<<(K_ * H_) / 256, 256, 0, stream>>>(proj_w2, pw2, K_, H_);
    gemm_gelu_pa_kernel<<<gemm_blocks((int)PAIRS, K_), 128, 0, stream>>>(
        g, pw1, proj_b1, hproj, K_, K_);
    gemm_biasperm_kernel<<<gemm_blocks((int)PAIRS, H_), 128, 0, stream>>>(
        hproj, pw2, proj_b2, biasA);

    // --- token embedding + LN -> x ---
    embed_ln_kernel<<<ROWS / 4, 128, 0, stream>>>(src_tokens, token_embed,
                                                  emb_ln_s, emb_ln_b, x);

    // --- transformer layers ---
    for (int l = 0; l < L_; ++l) {
        pack_b_kernel<<<(D_ * D_) / 256, 256, 0, stream>>>(Wq + (size_t)l * D_ * D_, pq, D_, D_);
        pack_b_kernel<<<(D_ * D_) / 256, 256, 0, stream>>>(Wk + (size_t)l * D_ * D_, pk, D_, D_);
        pack_b_kernel<<<(D_ * D_) / 256, 256, 0, stream>>>(Wv + (size_t)l * D_ * D_, pv, D_, D_);
        pack_b_kernel<<<(D_ * D_) / 256, 256, 0, stream>>>(Wo + (size_t)l * D_ * D_, po, D_, D_);
        pack_b_kernel<<<(D_ * F_) / 256, 256, 0, stream>>>(Wf1 + (size_t)l * D_ * F_, pf1, D_, F_);
        pack_b_kernel<<<(F_ * D_) / 256, 256, 0, stream>>>(Wf2 + (size_t)l * F_ * D_, pf2, F_, D_);

        const float* bias_in  = (l & 1) ? biasB : biasA;
        float*       bias_out = (l & 1) ? biasA : biasB;

        // pre-LN attention
        ln_pa_kernel<<<ROWS / 4, 128, 0, stream>>>(x, ln1_s + l * D_, ln1_b + l * D_, y);
        gemm_f32_kernel<<<gemm_blocks(ROWS, D_), 128, 0, stream>>>(
            y, pq, bq + l * D_, qb, D_, D_, scale);
        gemm_kv_perm_kernel<<<gemm_blocks(ROWS, D_), 128, 0, stream>>>(
            y, pk, bk + l * D_, kT);
        gemm_kv_perm_kernel<<<gemm_blocks(ROWS, D_), 128, 0, stream>>>(
            y, pv, bv + l * D_, vT);
        scores_softmax_kernel<<<(B_ * H_ * N_) / 4, 128, 0, stream>>>(
            qb, kT, bias_in, bias_out, probs);           // bias_out = this layer's scores
        attn_out_kernel<<<(B_ * N_ * D_) / 256, 256, 0, stream>>>(probs, vT, ob);
        gemm_res_f32_kernel<<<gemm_blocks(ROWS, D_), 128, 0, stream>>>(
            ob, po, bo + l * D_, x, D_, D_);

        // pre-LN FFN
        ln_pa_kernel<<<ROWS / 4, 128, 0, stream>>>(x, ln2_s + l * D_, ln2_b + l * D_, y);
        gemm_gelu_pa_kernel<<<gemm_blocks(ROWS, F_), 128, 0, stream>>>(
            y, pf1, bf1 + l * F_, h1, F_, D_);
        gemm_res_f32_kernel<<<gemm_blocks(ROWS, D_), 128, 0, stream>>>(
            h1, pf2, bf2 + l * D_, x, D_, F_);
    }

    // --- final LN -> d_out (fp32) ---
    ln_f32_kernel<<<ROWS / 4, 128, 0, stream>>>(x, fin_ln_s, fin_ln_b, (float*)d_out);

    (void)in_sizes; (void)n_in; (void)out_size; (void)ws_size;
}
